// Structure_Decoder_81131932221579
// MI455X (gfx1250) — matrix operand order repaired and checked by reading, not compiled, not run on hardware
//
#include <hip/hip_runtime.h>
#include <hip/hip_bf16.h>

typedef __attribute__((ext_vector_type(16))) _Float16 v16h;
typedef __attribute__((ext_vector_type(8)))  _Float16 v8h;
typedef __attribute__((ext_vector_type(8)))  float    v8f;
typedef __attribute__((ext_vector_type(4)))  float    v4f;
typedef __attribute__((ext_vector_type(2)))  float    v2f;

#define NH 128  // NHID is 128 in the reference; hardcode for unrolling
#define JB 8    // j-tiles per wave in the big GEMM (A-operand reuse factor)

// ---------------- degree / normalization ----------------

__global__ void k_deg_init(float* __restrict__ deg, int n) {
  int i = blockIdx.x * blockDim.x + threadIdx.x;
  if (i < n) deg[i] = 1.0f;  // self-loop contributes 1
}

__global__ void k_deg_acc(const int* __restrict__ col, float* __restrict__ deg, int e) {
  int i = blockIdx.x * blockDim.x + threadIdx.x;
  if (i < e) {
    int c = __builtin_nontemporal_load(col + i);  // edge list is streamed once
    atomicAdd(&deg[c], 1.0f);
  }
}

__global__ void k_dinv(const float* __restrict__ deg, float* __restrict__ dinv, int n) {
  int i = blockIdx.x * blockDim.x + threadIdx.x;
  if (i < n) {
    float d = deg[i];
    dinv[i] = (d > 0.0f) ? rsqrtf(d) : 0.0f;
  }
}

// ---------------- xw = x @ W  (FP32 WMMA 16x16x4, full precision) ----------------
// One wave per 16x16 output tile. K loop of 128 in steps of 4.
// A 16x4 f32: lanes 0-15 -> M=lane, V0=K0,V1=K1 ; lanes 16-31 -> V0=K2,V1=K3
// B 4x16 f32: lanes 0-15 -> N=lane, V0=K0,V1=K1 ; lanes 16-31 -> V0=K2,V1=K3
// C/D 16x16 f32: VGPR r -> M = r (+8 upper half-wave), N = lane%16
__global__ void k_xw_wmma(const float* __restrict__ x, const float* __restrict__ w,
                          float* __restrict__ xw, int n) {
  const int lane = threadIdx.x & 31;
  const int wave = (blockIdx.x * blockDim.x + threadIdx.x) >> 5;
  const int tilesJ = NH / 16;                 // 8
  const int tilesI = n >> 4;
  if (wave >= tilesI * tilesJ) return;        // wave-uniform: EXEC stays all-1s
  const int ti = wave / tilesJ;
  const int tj = wave - ti * tilesJ;
  const int hi = lane >> 4;                   // half-wave select
  const int l  = lane & 15;
  const int rowA = ti * 16 + l;
  const int colB = tj * 16 + l;

  v8f acc = {};
#pragma unroll 4
  for (int k0 = 0; k0 < NH; k0 += 4) {
    v2f a = *(const v2f*)(x + (size_t)rowA * NH + k0 + hi * 2);   // K = k0+2*hi, +1
    v2f b;
    b[0] = w[(size_t)(k0 + hi * 2 + 0) * NH + colB];
    b[1] = w[(size_t)(k0 + hi * 2 + 1) * NH + colB];
    acc = __builtin_amdgcn_wmma_f32_16x16x4_f32(
        false, a, false, b, (short)0, acc, false, false);
  }
#pragma unroll
  for (int r = 0; r < 8; ++r)
    xw[(size_t)(ti * 16 + r + hi * 8) * NH + colB] = acc[r];
}

// ---------------- aggregation ----------------

// self-loop term: agg[i,:] = dinv[i]^2 * xw[i,:]
__global__ void k_agg_init(const float* __restrict__ xw, const float* __restrict__ dinv,
                           float* __restrict__ agg, int n) {
  int i = blockIdx.x * blockDim.x + threadIdx.x;
  if (i < n * NH) {
    float d = dinv[i >> 7];
    agg[i] = d * d * xw[i];
  }
}

// one wave per edge; each lane handles 4 features via f32 L2 atomics
__global__ void k_edge_scatter(const int* __restrict__ row, const int* __restrict__ col,
                               const float* __restrict__ xw, const float* __restrict__ dinv,
                               float* __restrict__ agg, int e) {
  int t = blockIdx.x * blockDim.x + threadIdx.x;
  int edge = t >> 5;
  int lane = t & 31;
  if (edge >= e) return;
  int r = __builtin_nontemporal_load(row + edge);
  int c = __builtin_nontemporal_load(col + edge);
  float norm = dinv[r] * dinv[c];
  v4f m = *(const v4f*)(xw + (size_t)r * NH + lane * 4);
  float* dst = agg + (size_t)c * NH + lane * 4;
  atomicAdd(dst + 0, norm * m[0]);
  atomicAdd(dst + 1, norm * m[1]);
  atomicAdd(dst + 2, norm * m[2]);
  atomicAdd(dst + 3, norm * m[3]);
}

// h = relu(agg + b) downcast to f16 for the big WMMA GEMM
__global__ void k_relu_h16(const float* __restrict__ agg, const float* __restrict__ b,
                           _Float16* __restrict__ h16, int n) {
  int i = blockIdx.x * blockDim.x + threadIdx.x;
  if (i < n * NH) {
    float v = agg[i] + b[i & (NH - 1)];
    h16[i] = (_Float16)fmaxf(v, 0.0f);
  }
}

// ---------------- OUT = h @ h^T  (F16 WMMA 16x16x32, f32 accum) ----------------
// One wave computes a 1xJB strip of 16x16 tiles: the A operand (16x32 chunk of
// h rows) is loaded once per K-chunk and reused by JB WMMAs. h16 (2.56 MB)
// stays hot in the 192 MB L2; the 400 MB output streams to HBM with NT stores
// so it cannot thrash L2. Store bandwidth (23.3 TB/s -> ~17 us) is the floor;
// f16 WMMA compute (~25.6 GFLOP) sits well under it.
// A 16x32 f16 layout: lane m, V0-7 -> K = hi*16 + 0..15 (contiguous, matches B)
// B 32x16 f16 layout: lane n,  V0-7  -> K = hi*16 + 0..15 (contiguous -> b128 loads)
__global__ void k_hht_wmma(const _Float16* __restrict__ h, float* __restrict__ out, int n) {
  const int lane = threadIdx.x & 31;
  const int wave = (blockIdx.x * blockDim.x + threadIdx.x) >> 5;
  const int tilesI  = n >> 4;                       // 625
  const int groupsJ = (tilesI + JB - 1) / JB;       // 79
  if (wave >= tilesI * groupsJ) return;             // wave-uniform
  const int ti = wave / groupsJ;
  const int tg = wave - ti * groupsJ;
  const int hi = lane >> 4;
  const int l  = lane & 15;
  const int rowA = ti * 16 + l;
  const int jt0  = tg * JB;
  const int nj   = (tilesI - jt0 < JB) ? (tilesI - jt0) : JB;   // wave-uniform

  v8f acc[JB];
#pragma unroll
  for (int j = 0; j < JB; ++j) acc[j] = (v8f){};

#pragma unroll
  for (int k0 = 0; k0 < NH; k0 += 32) {
    // A operand: one contiguous 32B load per lane, reused across JB tiles
    v16h a = *(const v16h*)(h + (size_t)rowA * NH + k0 + hi * 16);

#pragma unroll
    for (int j = 0; j < JB; ++j) {
      if (j < nj) {                                 // uniform branch, EXEC all-1s
        const int rowB = (jt0 + j) * 16 + l;
        v16h b = *(const v16h*)(h + (size_t)rowB * NH + k0 + hi * 16);
        acc[j] = __builtin_amdgcn_wmma_f32_16x16x32_f16(
            false, a, false, b, (short)0, acc[j], false, false);
      }
    }
  }

#pragma unroll
  for (int j = 0; j < JB; ++j) {
    if (j < nj) {
      const int jc = (jt0 + j) * 16 + l;
#pragma unroll
      for (int r = 0; r < 8; ++r) {
        float* p = out + (size_t)(ti * 16 + r + hi * 8) * (size_t)n + jc;
        __builtin_nontemporal_store(acc[j][r], p);  // stream 400 MB past L2
      }
    }
  }
}

// ---------------- launch ----------------

extern "C" void kernel_launch(void* const* d_in, const int* in_sizes, int n_in,
                              void* d_out, int out_size, void* d_ws, size_t ws_size,
                              hipStream_t stream) {
  const float* x  = (const float*)d_in[0];
  const int*   ei = (const int*)d_in[1];     // [2,E] flat: first E = row(src), next E = col(dst)
  const float* W  = (const float*)d_in[2];
  const float* b  = (const float*)d_in[3];
  float* out = (float*)d_out;

  const int nh = in_sizes[3];                // 128
  const int n  = in_sizes[0] / nh;           // 10000
  const int e  = in_sizes[1] / 2;            // 320000
  const int* row = ei;
  const int* col = ei + e;

  // workspace layout (256B-aligned regions)
  char* ws = (char*)d_ws;
  size_t featBytes = (size_t)n * NH * sizeof(float);        // 5.12 MB
  float*    xw   = (float*)(ws);
  float*    agg  = (float*)(ws + featBytes);
  float*    deg  = (float*)(ws + 2 * featBytes);
  size_t nPad = ((size_t)n * sizeof(float) + 255) & ~(size_t)255;
  float*    dinv = (float*)(ws + 2 * featBytes + nPad);
  _Float16* h16  = (_Float16*)(ws + 2 * featBytes + 2 * nPad);

  const int B = 256;

  k_deg_init<<<(n + B - 1) / B, B, 0, stream>>>(deg, n);
  k_deg_acc<<<(e + B - 1) / B, B, 0, stream>>>(col, deg, e);
  k_dinv<<<(n + B - 1) / B, B, 0, stream>>>(deg, dinv, n);

  {
    int waves = (n >> 4) * (NH / 16);
    int thr = waves * 32;
    k_xw_wmma<<<(thr + B - 1) / B, B, 0, stream>>>(x, W, xw, n);
  }

  k_agg_init<<<((size_t)n * NH + B - 1) / B, B, 0, stream>>>(xw, dinv, agg, n);

  {
    long long thr = (long long)e * 32;
    k_edge_scatter<<<(int)((thr + B - 1) / B), B, 0, stream>>>(row, col, xw, dinv, agg, e);
  }

  k_relu_h16<<<((size_t)n * NH + B - 1) / B, B, 0, stream>>>(agg, b, h16, n);

  {
    int tilesI  = n >> 4;
    int groupsJ = (tilesI + JB - 1) / JB;
    long long waves = (long long)tilesI * groupsJ;
    long long thr = waves * 32;
    k_hht_wmma<<<(int)((thr + B - 1) / B), B, 0, stream>>>(h16, out, n);
  }
}
